// ConvLSTM_18287970746516
// MI455X (gfx1250) — matrix operand order repaired and checked by reading, not compiled, not run on hardware
//
#include <hip/hip_runtime.h>

// ---------------------------------------------------------------------------
// ConvLSTM (N=256, DIN=256, DH=512, 8x8 spatial, 3x3 reflect-pad convs)
// Implicit-GEMM on CDNA5 WMMA f16 (16x16x32, f32 accum), TDM-staged LDS.
//   M = 64 positions per image (4 tiles of 16)
//   N = 2048 output channels (= 4 gates x 512)
//   K = 9*256 (x-conv) + 9*512 (h-conv) = 6912, chunked by 32
// One workgroup = 1 image x 32 DH-channels x all 4 gates (8 waves), so the
// LSTM gate math fuses in-kernel via an LDS pre-activation exchange.
// Staging: pre-pass materializes reflect-padded f16 planes in global memory;
// the hot kernel pulls them into LDS with tensor_load_to_lds (TDM), using
// pad_enable to insert the bank-conflict-breaking LDS row pitch.
// ---------------------------------------------------------------------------

typedef __attribute__((ext_vector_type(16))) _Float16 v16h;
typedef __attribute__((ext_vector_type(8)))  _Float16 v8h;
typedef __attribute__((ext_vector_type(8)))  float    v8f;
typedef __attribute__((ext_vector_type(4)))  unsigned int v4u;
typedef __attribute__((ext_vector_type(8)))  int v8i;
typedef __attribute__((ext_vector_type(4)))  int v4i;

#define NBATCH 256
#define DIN    256
#define DH     512
#define KX     2304            // 9*DIN
#define KH     4608            // 9*DH
#define XP_PITCH 264           // 256 + 8 pad halves (LDS bank-conflict break)
#define HP_PITCH 520           // 512 + 8 pad halves
#define XS_HALF  0
#define HS_HALF  (100*XP_PITCH)               // 26400 halves = 52800 bytes
#define PRE_BYTE ((HS_HALF + 100*HP_PITCH)*2) // 156800 bytes
#define LDS_BYTES (PRE_BYTE + 4*32*64*4)      // +32KB pre-act exchange = 189568

__device__ __forceinline__ int reflect8(int v) {
  v -= 1; if (v < 0) v = -v; if (v > 7) v = 14 - v; return v;
}
__device__ __forceinline__ float sigmoidf_(float x) {
  return 1.0f / (1.0f + expf(-x));
}

// ---------------------------------------------------------------------------
// Pack f32 OIHW weights into f16 B-matrices, layout B[o][k], k = (ky*3+kx)*C+c
// so each WMMA lane reads 16 contiguous halves of its column.
// ---------------------------------------------------------------------------
__global__ void pack_weights(const float* __restrict__ wii, const float* __restrict__ wif,
                             const float* __restrict__ wig, const float* __restrict__ wio,
                             const float* __restrict__ whi, const float* __restrict__ whf,
                             const float* __restrict__ whg, const float* __restrict__ who,
                             _Float16* __restrict__ Bx, _Float16* __restrict__ Bh)
{
  size_t i = (size_t)blockIdx.x * blockDim.x + threadIdx.x;
  const size_t NX = (size_t)2048 * KX;
  const size_t NH = (size_t)2048 * KH;
  if (i < NX) {
    int o = (int)(i / KX), k = (int)(i % KX);
    int kyx = k >> 8, c = k & 255;
    const float* w = (o < 512) ? wii : (o < 1024) ? wif : (o < 1536) ? wig : wio;
    Bx[i] = (_Float16)w[(size_t)(o & 511) * (DIN*9) + c*9 + kyx];
  } else if (i < NX + NH) {
    size_t j = i - NX;
    int o = (int)(j / KH), k = (int)(j % KH);
    int kyx = k >> 9, c = k & 511;
    const float* w = (o < 512) ? whi : (o < 1024) ? whf : (o < 1536) ? whg : who;
    Bh[j] = (_Float16)w[(size_t)(o & 511) * (DH*9) + c*9 + kyx];
  }
}

// ---------------------------------------------------------------------------
// Materialize reflect-padded f16 image planes: xpad[n][pos100][c256],
// hpad[n][pos100][c512] (dense rows; TDM adds the LDS pitch at copy time).
// ---------------------------------------------------------------------------
__global__ void pad_inputs(const float* __restrict__ xin, const float* __restrict__ hin,
                           _Float16* __restrict__ xpad, _Float16* __restrict__ hpad)
{
  size_t i = (size_t)blockIdx.x * blockDim.x + threadIdx.x;
  const size_t NXP = (size_t)NBATCH * 100 * DIN;
  const size_t NHP = (size_t)NBATCH * 100 * DH;
  if (i < NXP) {
    int c   = (int)(i % DIN);
    int pos = (int)((i / DIN) % 100);
    int n   = (int)(i / ((size_t)DIN * 100));
    int py = pos / 10, px = pos - py*10;
    xpad[i] = (_Float16)xin[(((size_t)n*DIN + c) << 6) + (reflect8(py) << 3) + reflect8(px)];
  } else if (i < NXP + NHP) {
    size_t j = i - NXP;
    int c   = (int)(j % DH);
    int pos = (int)((j / DH) % 100);
    int n   = (int)(j / ((size_t)DH * 100));
    int py = pos / 10, px = pos - py*10;
    hpad[j] = (_Float16)hin[(size_t)n*(2*DH*64) + ((size_t)c << 6) + (reflect8(py) << 3) + reflect8(px)];
  }
}

// ---------------------------------------------------------------------------
// Fused implicit-GEMM ConvLSTM cell.  grid = (DH/32, N), block = 256 (8 waves)
// ---------------------------------------------------------------------------
__global__ __launch_bounds__(256, 1)
void convlstm_fused(const _Float16* __restrict__ xpad, const _Float16* __restrict__ hpad,
                    const float* __restrict__ hin,
                    const float* __restrict__ b_i, const float* __restrict__ b_f,
                    const float* __restrict__ b_g, const float* __restrict__ b_o,
                    const float* __restrict__ Wci, const float* __restrict__ Wcf,
                    const float* __restrict__ Wco,
                    const _Float16* __restrict__ Bx, const _Float16* __restrict__ Bh,
                    float* __restrict__ out)
{
  extern __shared__ __align__(32) char smem[];
  _Float16* lh  = (_Float16*)smem;
  float*    pre = (float*)(smem + PRE_BYTE);

  const int n   = blockIdx.y;
  const int d0  = blockIdx.x << 5;     // 32 DH channels per block
  const int tid = threadIdx.x;

  // ---- TDM: DMA both padded planes into LDS, inserting row pitch pad ----
  // D# group0: count=1 | lds_addr | global_addr | type=2
  // D# group1: data_size=2B, pad_enable, pad_interval(row DWORDs), pad_amount=4DW,
  //            tensor_dim0=row, tensor_dim1=100, tile=row x 100, stride0=row.
  if (tid < 32) {
    unsigned long long ga = (unsigned long long)(uintptr_t)(xpad + (size_t)n * 100 * DIN);
    v4u g0x = { 1u, (unsigned)(XS_HALF * 2), (unsigned)(ga & 0xffffffffu),
                (unsigned)((ga >> 32) & 0x1ffffffu) | (2u << 30) };
    v8i g1x = { (int)((1u << 16) | (1u << 20) | (6u << 22) | (3u << 25)),
                (int)(256u << 16), (int)(100u << 16), (int)(256u << 16),
                100, 256, 0, 0 };
    v4i gz4 = { 0, 0, 0, 0 };
    v8i gz8 = { 0, 0, 0, 0, 0, 0, 0, 0 };
    __builtin_amdgcn_tensor_load_to_lds(g0x, g1x, gz4, gz4, gz8, 0);

    unsigned long long gb = (unsigned long long)(uintptr_t)(hpad + (size_t)n * 100 * DH);
    v4u g0h = { 1u, (unsigned)(HS_HALF * 2), (unsigned)(gb & 0xffffffffu),
                (unsigned)((gb >> 32) & 0x1ffffffu) | (2u << 30) };
    v8i g1h = { (int)((1u << 16) | (1u << 20) | (7u << 22) | (3u << 25)),
                (int)(512u << 16), (int)(100u << 16), (int)(512u << 16),
                100, 512, 0, 0 };
    __builtin_amdgcn_tensor_load_to_lds(g0h, g1h, gz4, gz4, gz8, 0);

    __builtin_amdgcn_s_wait_tensorcnt(0);
  }
  __syncthreads();

  const int wave  = tid >> 5;
  const int lane  = tid & 31;
  const int l16   = lane & 15;
  const int lhalf = lane >> 4;                 // 0: lanes 0-15, 1: lanes 16-31
  const int gate  = wave >> 1;                 // i,f,g,o
  const int n0    = gate*DH + d0 + ((wave & 1) << 4);   // wave's 16 channels

  const _Float16* bx_lane = Bx + (size_t)(n0 + l16)*KX + (lhalf << 4);
  const _Float16* bh_lane = Bh + (size_t)(n0 + l16)*KH + (lhalf << 4);
  const int off1 = lhalf << 4;                 // A-frag K sub-offset (0 or 16)

  int My[4], Mx[4];
  #pragma unroll
  for (int m = 0; m < 4; ++m) { int M = (m << 4) + l16; My[m] = M >> 3; Mx[m] = M & 7; }

  v8f acc[4] = {};

  // One K=32 chunk: issue all 8 ds_load_b128 for the 4 A-fragments first,
  // then the 4 WMMAs (staggered dscnt waits hide LDS latency).
  // A contiguous K layout (matches B): lanes0-15 K{0-15}, lanes16-31 K{16-31}.
  // base/pitch/shift are literals at every call site -> constant-stride codegen.
  auto mma_body = [&](int q, v16h bfrag, int base, int pitch, int shift) {
    const int kyx = q >> shift;
    const int cb  = (q & ((1 << shift) - 1)) << 5;
    const int ky  = kyx / 3, kx = kyx - ky*3;
    v16h af[4];
    #pragma unroll
    for (int m = 0; m < 4; ++m) {
      const int pp = (My[m] + ky)*10 + Mx[m] + kx;    // padded position
      const _Float16* ap = lh + base + pp*pitch + cb + off1;
      v8h lo = *(const v8h*)ap;
      v8h hi = *(const v8h*)(ap + 8);
      #pragma unroll
      for (int i = 0; i < 8; ++i) { af[m][i] = lo[i]; af[m][8 + i] = hi[i]; }
    }
    #pragma unroll
    for (int m = 0; m < 4; ++m) {
      acc[m] = __builtin_amdgcn_wmma_f32_16x16x32_f16(
          false, af[m], false, bfrag, (short)0, acc[m], false, false);
    }
  };

  // ---- x-conv region: 72 chunks (8 per 3x3 tap), B prefetched 1 ahead ----
  v16h bf = *(const v16h*)bx_lane;
  #pragma unroll 1
  for (int q = 0; q < 71; ++q) {
    v16h bn = *(const v16h*)(bx_lane + ((q + 1) << 5));
    mma_body(q, bf, XS_HALF, XP_PITCH, 3);
    bf = bn;
  }
  {   // seam: last x chunk, prefetch first h chunk
    v16h bn = *(const v16h*)bh_lane;
    mma_body(71, bf, XS_HALF, XP_PITCH, 3);
    bf = bn;
  }
  // ---- h-conv region: 144 chunks (16 per 3x3 tap) ----
  #pragma unroll 1
  for (int q = 0; q < 143; ++q) {
    v16h bn = *(const v16h*)(bh_lane + ((q + 1) << 5));
    mma_body(q, bf, HS_HALF, HP_PITCH, 4);
    bf = bn;
  }
  mma_body(143, bf, HS_HALF, HP_PITCH, 4);

  // ---- exchange pre-activations through LDS: pre[gate][dh_local][pos] ----
  // C/D layout: VGPR r -> row M = r + 8*lhalf (position), col N = l16 (channel).
  #pragma unroll
  for (int m = 0; m < 4; ++m) {
    #pragma unroll
    for (int r = 0; r < 8; ++r) {
      int pos = (m << 4) + r + (lhalf << 3);
      pre[gate*2048 + ((((wave & 1) << 4) + l16) << 6) + pos] = acc[m][r];
    }
  }
  __syncthreads();

  // ---- fused LSTM gate math (32 dh x 64 pos per block) ----
  float* hout = out + (size_t)NBATCH * DH * 64;
  for (int e = tid; e < 32*64; e += 256) {
    int dhl = e >> 6, pos = e & 63;
    int dh  = d0 + dhl;
    float pi = pre[        (dhl << 6) + pos] + b_i[dh];
    float pf = pre[2048 +  (dhl << 6) + pos] + b_f[dh];
    float pg = pre[4096 +  (dhl << 6) + pos] + b_g[dh];
    float po = pre[6144 +  (dhl << 6) + pos] + b_o[dh];
    int   wi = (dh << 6) + pos;
    float c0 = hin[(size_t)n*(2*DH*64) + DH*64 + wi];
    float ig = sigmoidf_(pi + c0 * Wci[wi]);
    float fg = sigmoidf_(pf + c0 * Wcf[wi]);
    float gg = tanhf(pg);
    float ct = fg * c0 + ig * gg;
    float og = sigmoidf_(po + ct * Wco[wi]);
    float ht = og * tanhf(ct);
    out [(size_t)n*(DH*64) + wi]           = og;   // output o
    hout[(size_t)n*(2*DH*64) + wi]         = ht;   // hidden_state_out[:,0]
    hout[(size_t)n*(2*DH*64) + DH*64 + wi] = ct;   // hidden_state_out[:,1]
  }
}

extern "C" void kernel_launch(void* const* d_in, const int* in_sizes, int n_in,
                              void* d_out, int out_size, void* d_ws, size_t ws_size,
                              hipStream_t stream) {
  const float* xin = (const float*)d_in[0];
  const float* hin = (const float*)d_in[1];
  const float* wii = (const float*)d_in[2];
  const float* wif = (const float*)d_in[3];
  const float* wig = (const float*)d_in[4];
  const float* wio = (const float*)d_in[5];
  const float* whi = (const float*)d_in[6];
  const float* whf = (const float*)d_in[7];
  const float* whg = (const float*)d_in[8];
  const float* who = (const float*)d_in[9];
  const float* b_i = (const float*)d_in[10];
  const float* b_f = (const float*)d_in[11];
  const float* b_g = (const float*)d_in[12];
  const float* b_o = (const float*)d_in[13];
  const float* Wci = (const float*)d_in[14];
  const float* Wcf = (const float*)d_in[15];
  const float* Wco = (const float*)d_in[16];
  float* out = (float*)d_out;

  // Workspace layout
  _Float16* Bx   = (_Float16*)d_ws;                       //  9,437,184 B
  _Float16* Bh   = Bx   + (size_t)2048 * KX;              // 18,874,368 B
  _Float16* xpad = Bh   + (size_t)2048 * KH;              // 13,107,200 B
  _Float16* hpad = xpad + (size_t)NBATCH * 100 * DIN;     // 26,214,400 B

  const size_t wtot = (size_t)2048 * KX + (size_t)2048 * KH;
  pack_weights<<<(unsigned)((wtot + 255) / 256), 256, 0, stream>>>(
      wii, wif, wig, wio, whi, whf, whg, who, Bx, Bh);

  const size_t ptot = (size_t)NBATCH * 100 * (DIN + DH);
  pad_inputs<<<(unsigned)((ptot + 255) / 256), 256, 0, stream>>>(xin, hin, xpad, hpad);

  (void)hipFuncSetAttribute((const void*)convlstm_fused,
                            hipFuncAttributeMaxDynamicSharedMemorySize, LDS_BYTES);
  dim3 grid(DH / 32, NBATCH);
  convlstm_fused<<<grid, 256, LDS_BYTES, stream>>>(
      xpad, hpad, hin, b_i, b_f, b_g, b_o, Wci, Wcf, Wco, Bx, Bh, out);

  (void)in_sizes; (void)n_in; (void)out_size; (void)ws_size;
}
